// CausalSelfAttention_43044162240980
// MI455X (gfx1250) — matrix operand order repaired and checked by reading, not compiled, not run on hardware
//
#include <hip/hip_runtime.h>
#include <hip/hip_bf16.h>

typedef __bf16 bf16_t;
typedef __attribute__((ext_vector_type(16))) __bf16 v16bf;
typedef __attribute__((ext_vector_type(8)))  __bf16 v8bf;
typedef __attribute__((ext_vector_type(8)))  float  v8f;

#define BATCH 2
#define LSEQ  2048
#define DIM   1024
#define NHEAD 16
#define HD    64
#define MROWS (BATCH*LSEQ)   // 4096

__device__ __forceinline__ v8f wmma_bf16(v16bf a, v16bf b, v8f c) {
  return __builtin_amdgcn_wmma_f32_16x16x32_bf16(false, a, false, b, (short)0, c,
                                                 false, false);
}

// Async global -> LDS copy, 16B per lane. lds_off = LDS byte address (low 32
// bits of a generic pointer into __shared__), gptr = global address.
__device__ __forceinline__ void async_b128(unsigned lds_off, const void* gptr) {
  asm volatile("global_load_async_to_lds_b128 %0, %1, off"
               :: "v"(lds_off), "v"(gptr) : "memory");
}

// A-fragment (M=16 x K=32, bf16, row-major src, row stride ld elements).
// Lane r = lane%16 holds row r; K elems {ko..ko+7} and {16+ko..16+ko+7},
// ko = (lane<16 ? 0 : 8).
__device__ __forceinline__ v16bf load_frag_a(const bf16_t* src, int ld, int lane) {
  int r  = lane & 15;
  int ko = (lane < 16) ? 0 : 8;
  const bf16_t* p = src + (size_t)r * ld + ko;
  union { v16bf v; v8bf h[2]; } u;
  u.h[0] = *(const v8bf*)(p);
  u.h[1] = *(const v8bf*)(p + 16);
  return u.v;
}

// B-fragment (K=32 x N=16), stored via its transpose: src = N x K row-major
// rows (W rows / K rows / V^T rows). Lane n = lane%16 holds column n; K elems
// {ko..ko+7} and {16+ko..16+ko+7}, ko = (lane<16 ? 0 : 8).
__device__ __forceinline__ v16bf load_frag_b(const bf16_t* src, int ld, int lane) {
  int n  = lane & 15;
  int ko = (lane < 16) ? 0 : 8;
  const bf16_t* p = src + (size_t)n * ld + ko;
  union { v16bf v; v8bf h[2]; } u;
  u.h[0] = *(const v8bf*)(p);
  u.h[1] = *(const v8bf*)(p + 16);
  return u.v;
}

// ---------------------------------------------------------------- convert ---
__global__ __launch_bounds__(256) void cvt_f32_bf16(const float* __restrict__ in,
                                                    bf16_t* __restrict__ out, int n) {
  int i = (blockIdx.x * blockDim.x + threadIdx.x) * 4;
  if (i < n) {
    float4 f = *(const float4*)(in + i);
    out[i + 0] = (bf16_t)f.x;
    out[i + 1] = (bf16_t)f.y;
    out[i + 2] = (bf16_t)f.z;
    out[i + 3] = (bf16_t)f.w;
  }
}

// ------------------------------------------------ shared GEMM main loop ----
// Block = 256 threads (8 waves) computes a 128x128 tile of y = A @ W^T.
// A and W panels (128 rows x 32 k, bf16) are double-buffered in LDS via
// global_load_async_to_lds_b128. Wave (wr,wc) in a 4x2 grid owns 32x64.
// LDS rows padded to 40 bf16 (80B) to spread ds_load_b128 banks.
#define KPAD 40
#define PANEL_B (128 * KPAD * 2)   // bytes per panel buffer

__device__ __forceinline__ void gemm_mainloop(const bf16_t* __restrict__ A,
                                              const bf16_t* __restrict__ W,
                                              int row0, int col0, v8f acc[2][4]) {
  __shared__ __align__(16) bf16_t sA[2][128][KPAD];
  __shared__ __align__(16) bf16_t sB[2][128][KPAD];

  int tid  = threadIdx.x;
  int lane = tid & 31;
  int wave = tid >> 5;
  int wr   = wave >> 1;   // 0..3: 32-row group
  int wc   = wave & 1;    // 0..1: 64-col group

  unsigned sAbase = (unsigned)(uintptr_t)&sA[0][0][0];
  unsigned sBbase = (unsigned)(uintptr_t)&sB[0][0][0];

  // Each thread owns two 16B chunks of each panel per stage (512 chunks total).
  int g0 = tid, g1 = tid + 256;
  int r0 = g0 >> 2, p0 = g0 & 3;
  int r1 = g1 >> 2, p1 = g1 & 3;

  auto issue = [&](int buf, int k0) {
    const bf16_t* Ab = A + (size_t)row0 * DIM + k0;
    const bf16_t* Wb = W + (size_t)col0 * DIM + k0;
    unsigned la = sAbase + buf * PANEL_B;
    unsigned lb = sBbase + buf * PANEL_B;
    async_b128(la + r0 * (KPAD * 2) + p0 * 16, Ab + (size_t)r0 * DIM + p0 * 8);
    async_b128(la + r1 * (KPAD * 2) + p1 * 16, Ab + (size_t)r1 * DIM + p1 * 8);
    async_b128(lb + r0 * (KPAD * 2) + p0 * 16, Wb + (size_t)r0 * DIM + p0 * 8);
    async_b128(lb + r1 * (KPAD * 2) + p1 * 16, Wb + (size_t)r1 * DIM + p1 * 8);
  };

  issue(0, 0);
  for (int ks = 0; ks < DIM / 32; ++ks) {
    if (ks + 1 < DIM / 32) {
      issue((ks + 1) & 1, (ks + 1) * 32);
      asm volatile("s_wait_asynccnt 0x4" ::: "memory");  // stage ks done
    } else {
      asm volatile("s_wait_asynccnt 0x0" ::: "memory");
    }
    __syncthreads();
    const bf16_t* bA = &sA[ks & 1][0][0];
    const bf16_t* bB = &sB[ks & 1][0][0];
    v16bf a0 = load_frag_a(bA + (wr * 32 +  0) * KPAD, KPAD, lane);
    v16bf a1 = load_frag_a(bA + (wr * 32 + 16) * KPAD, KPAD, lane);
#pragma unroll
    for (int t = 0; t < 4; ++t) {
      v16bf b = load_frag_b(bB + (wc * 64 + t * 16) * KPAD, KPAD, lane);
      acc[0][t] = wmma_bf16(a0, b, acc[0][t]);
      acc[1][t] = wmma_bf16(a1, b, acc[1][t]);
    }
    __syncthreads();
  }
}

// -------------------------------------------------------------- QKV GEMM ---
__global__ __launch_bounds__(256) void gemm_qkv(
    const bf16_t* __restrict__ X,
    const bf16_t* __restrict__ Wqb, const bf16_t* __restrict__ Wkb,
    const bf16_t* __restrict__ Wvb,
    const float* __restrict__ bq, const float* __restrict__ bk,
    const float* __restrict__ bv,
    bf16_t* __restrict__ Qo, bf16_t* __restrict__ Ko, bf16_t* __restrict__ Vt) {
  int z    = blockIdx.z;
  const bf16_t* W    = (z == 0) ? Wqb : (z == 1) ? Wkb : Wvb;
  const float*  bias = (z == 0) ? bq  : (z == 1) ? bk  : bv;
  int row0 = blockIdx.y * 128;
  int col0 = blockIdx.x * 128;

  v8f acc[2][4] = {};
  gemm_mainloop(X, W, row0, col0, acc);

  int lane = threadIdx.x & 31;
  int wave = threadIdx.x >> 5;
  int wr = wave >> 1, wc = wave & 1;
  int half = lane >> 4, jc = lane & 15;
#pragma unroll
  for (int p = 0; p < 2; ++p) {
#pragma unroll
    for (int t = 0; t < 4; ++t) {
      int   n   = col0 + wc * 64 + t * 16 + jc;
      float bv_ = bias[n];
      int   h   = n >> 6, d = n & 63;
#pragma unroll
      for (int v = 0; v < 8; ++v) {
        int m = row0 + wr * 32 + p * 16 + v + half * 8;
        int b = m >> 11, l = m & (LSEQ - 1);
        float val = acc[p][t][v] + bv_;
        if (z == 0)
          Qo[((size_t)(b * NHEAD + h) * LSEQ + l) * HD + d] = (bf16_t)(val * 0.125f);
        else if (z == 1)
          Ko[((size_t)(b * NHEAD + h) * LSEQ + l) * HD + d] = (bf16_t)val;
        else
          Vt[((size_t)(b * NHEAD + h) * HD + d) * LSEQ + l] = (bf16_t)val;
      }
    }
  }
}

// ----------------------------------------------------- output projection ---
__global__ __launch_bounds__(256) void gemm_out(
    const bf16_t* __restrict__ A, const bf16_t* __restrict__ W,
    const float* __restrict__ bias, float* __restrict__ out) {
  int row0 = blockIdx.y * 128;
  int col0 = blockIdx.x * 128;

  v8f acc[2][4] = {};
  gemm_mainloop(A, W, row0, col0, acc);

  int lane = threadIdx.x & 31;
  int wave = threadIdx.x >> 5;
  int wr = wave >> 1, wc = wave & 1;
  int half = lane >> 4, jc = lane & 15;
#pragma unroll
  for (int p = 0; p < 2; ++p) {
#pragma unroll
    for (int t = 0; t < 4; ++t) {
      float bv = bias[col0 + wc * 64 + t * 16 + jc];
#pragma unroll
      for (int v = 0; v < 8; ++v) {
        int m = row0 + wr * 32 + p * 16 + v + half * 8;
        out[(size_t)m * DIM + col0 + wc * 64 + t * 16 + jc] = acc[p][t][v] + bv;
      }
    }
  }
}

// -------------------------------------------------------- flash attention ---
// Block = 8 waves covering 128 consecutive queries of one (b,h). K and V^T
// panels for each 32-kv chunk are staged once per block into LDS via async
// copies (double-buffered, s_wait_asynccnt), cutting global K/V traffic 8x.
// Waves whose causal range ends skip compute but keep the block barriers.
#define KROWP 72   // K panel: 32 rows x 64 d, padded (36-bank stride)
#define VROWP 40   // V panel: 64 rows x 32 kv, padded (20-bank stride)
#define KBUFB (32 * KROWP * 2)
#define VBUFB (64 * VROWP * 2)

__global__ __launch_bounds__(256) void flash_attn(
    const bf16_t* __restrict__ Q, const bf16_t* __restrict__ K,
    const bf16_t* __restrict__ Vt, bf16_t* __restrict__ O) {
  __shared__ __align__(16) bf16_t sKp[2][32][KROWP];   // 9216 B
  __shared__ __align__(16) bf16_t sVp[2][64][VROWP];   // 10240 B
  __shared__ __align__(16) bf16_t pbuf[8][16][32];     // 8192 B

  int tid  = threadIdx.x;
  int lane = tid & 31;
  int wave = tid >> 5;
  int bh   = blockIdx.x >> 4;            // b*NHEAD + h
  int qb   = (blockIdx.x & 15) * 128;    // block's first query row
  int q0   = qb + wave * 16;             // this wave's q-tile

  const bf16_t* Qh = Q  + (size_t)bh * LSEQ * HD;
  const bf16_t* Kh = K  + (size_t)bh * LSEQ * HD;
  const bf16_t* Vh = Vt + (size_t)bh * HD * LSEQ;

  unsigned sKbase = (unsigned)(uintptr_t)&sKp[0][0][0];
  unsigned sVbase = (unsigned)(uintptr_t)&sVp[0][0][0];

  int kr = tid >> 3, kp = tid & 7;       // K panel: 32 rows x 8 16B chunks
  int vr = tid >> 2, vp = tid & 3;       // V panel: 64 rows x 4 16B chunks

  auto issue = [&](int buf, int kv0) {
    async_b128(sKbase + buf * KBUFB + kr * (KROWP * 2) + kp * 16,
               Kh + (size_t)(kv0 + kr) * HD + kp * 8);
    async_b128(sVbase + buf * VBUFB + vr * (VROWP * 2) + vp * 16,
               Vh + (size_t)vr * LSEQ + kv0 + vp * 8);
  };

  v16bf qa0 = load_frag_a(Qh + (size_t)q0 * HD + 0,  HD, lane);
  v16bf qa1 = load_frag_a(Qh + (size_t)q0 * HD + 32, HD, lane);

  v8f   acc[4] = {};
  float mrow[8], lrow[8];
#pragma unroll
  for (int v = 0; v < 8; ++v) { mrow[v] = -1e30f; lrow[v] = 0.f; }

  int half = lane >> 4;
  int jc   = lane & 15;
  bf16_t* myp = &pbuf[wave][0][0];
  int nchunk = qb / 32 + 4;              // covers kv up to qb+127

  issue(0, 0);
  for (int c = 0; c < nchunk; ++c) {
    int kv0 = c * 32;
    if (c + 1 < nchunk) {
      issue((c + 1) & 1, kv0 + 32);
      asm volatile("s_wait_asynccnt 0x2" ::: "memory");   // chunk c landed
    } else {
      asm volatile("s_wait_asynccnt 0x0" ::: "memory");
    }
    __syncthreads();
    if (kv0 <= q0) {                     // chunk intersects causal range
      const bf16_t* bK = &sKp[c & 1][0][0];
      const bf16_t* bV = &sVp[c & 1][0][0];
#pragma unroll
      for (int sub = 0; sub < 2; ++sub) {
        int skv = kv0 + sub * 16;
        if (skv <= q0) {
          v16bf kb0 = load_frag_b(bK + (sub * 16) * KROWP + 0,  KROWP, lane);
          v16bf kb1 = load_frag_b(bK + (sub * 16) * KROWP + 32, KROWP, lane);
          v8f sc = {};
          sc = wmma_bf16(qa0, kb0, sc);
          sc = wmma_bf16(qa1, kb1, sc);
          if (skv == q0) {               // diagonal block: mask j > i
#pragma unroll
            for (int v = 0; v < 8; ++v) {
              int i = v + half * 8;
              if (jc > i) sc[v] = -1e30f;
            }
          }
#pragma unroll
          for (int v = 0; v < 8; ++v) {
            float mx = sc[v];
#pragma unroll
            for (int d = 1; d < 16; d <<= 1) mx = fmaxf(mx, __shfl_xor(mx, d, 32));
            float mn    = fmaxf(mrow[v], mx);
            float scale = __expf(mrow[v] - mn);
            mrow[v] = mn;
            float p  = __expf(sc[v] - mn);
            float rs = p;
#pragma unroll
            for (int d = 1; d < 16; d <<= 1) rs += __shfl_xor(rs, d, 32);
            lrow[v] = lrow[v] * scale + rs;
            acc[0][v] *= scale; acc[1][v] *= scale;
            acc[2][v] *= scale; acc[3][v] *= scale;
            myp[(v + half * 8) * 32 + sub * 16 + jc] = (bf16_t)p;
          }
        } else {                          // zero-pad second half of P
#pragma unroll
          for (int v = 0; v < 8; ++v)
            myp[(v + half * 8) * 32 + sub * 16 + jc] = (bf16_t)0.f;
        }
      }
      asm volatile("s_wait_dscnt 0x0" ::: "memory");
      v16bf pa = load_frag_a(myp, 32, lane);   // P in A-layout (16x32)
#pragma unroll
      for (int t = 0; t < 4; ++t) {            // O += P @ V  (B = V^T rows)
        v16bf vb = load_frag_b(bV + (size_t)(t * 16) * VROWP, VROWP, lane);
        acc[t] = wmma_bf16(pa, vb, acc[t]);
      }
    }
    __syncthreads();
  }

  int b = bh >> 4, h = bh & 15;
#pragma unroll
  for (int v = 0; v < 8; ++v) {
    float inv = 1.0f / lrow[v];
    int   l   = q0 + v + half * 8;
    size_t rowoff = ((size_t)b * LSEQ + l) * DIM + h * HD;
#pragma unroll
    for (int t = 0; t < 4; ++t)
      O[rowoff + t * 16 + jc] = (bf16_t)(acc[t][v] * inv);
  }
}

// ------------------------------------------------------------------ host ---
extern "C" void kernel_launch(void* const* d_in, const int* in_sizes, int n_in,
                              void* d_out, int out_size, void* d_ws, size_t ws_size,
                              hipStream_t stream) {
  const float* x  = (const float*)d_in[0];
  const float* Wq = (const float*)d_in[1];
  const float* bq = (const float*)d_in[2];
  const float* Wk = (const float*)d_in[3];
  const float* bk = (const float*)d_in[4];
  const float* Wv = (const float*)d_in[5];
  const float* bv = (const float*)d_in[6];
  const float* Wo = (const float*)d_in[7];
  const float* bo = (const float*)d_in[8];
  float* out = (float*)d_out;

  char* ws = (char*)d_ws;
  size_t off = 0;
  bf16_t* xb  = (bf16_t*)(ws + off); off += (size_t)MROWS * DIM * 2;      // 8 MB
  bf16_t* wqb = (bf16_t*)(ws + off); off += (size_t)DIM * DIM * 2;        // 2 MB
  bf16_t* wkb = (bf16_t*)(ws + off); off += (size_t)DIM * DIM * 2;
  bf16_t* wvb = (bf16_t*)(ws + off); off += (size_t)DIM * DIM * 2;
  bf16_t* wob = (bf16_t*)(ws + off); off += (size_t)DIM * DIM * 2;
  bf16_t* Qb  = (bf16_t*)(ws + off); off += (size_t)MROWS * DIM * 2;      // 8 MB
  bf16_t* Kb  = (bf16_t*)(ws + off); off += (size_t)MROWS * DIM * 2;
  bf16_t* Vtb = (bf16_t*)(ws + off); off += (size_t)MROWS * DIM * 2;
  bf16_t* Ob  = (bf16_t*)(ws + off); off += (size_t)MROWS * DIM * 2;

  // fp32 -> bf16
  int nx = MROWS * DIM, nw = DIM * DIM;
  cvt_f32_bf16<<<nx / 4 / 256, 256, 0, stream>>>(x,  xb,  nx);
  cvt_f32_bf16<<<nw / 4 / 256, 256, 0, stream>>>(Wq, wqb, nw);
  cvt_f32_bf16<<<nw / 4 / 256, 256, 0, stream>>>(Wk, wkb, nw);
  cvt_f32_bf16<<<nw / 4 / 256, 256, 0, stream>>>(Wv, wvb, nw);
  cvt_f32_bf16<<<nw / 4 / 256, 256, 0, stream>>>(Wo, wob, nw);

  // QKV projection: 128x128 tiles
  gemm_qkv<<<dim3(DIM / 128, MROWS / 128, 3), 256, 0, stream>>>(
      xb, wqb, wkb, wvb, bq, bk, bv, Qb, Kb, Vtb);

  // flash attention: 512 blocks, each 128 queries of one (b,h)
  flash_attn<<<BATCH * NHEAD * (LSEQ / 128), 256, 0, stream>>>(Qb, Kb, Vtb, Ob);

  // output projection -> fp32
  gemm_out<<<dim3(DIM / 128, MROWS / 128), 256, 0, stream>>>(Ob, wob, bo, out);
}
